// myPNA_37641093382882
// MI455X (gfx1250) — matrix operand order repaired and checked by reading, not compiled, not run on hardware
//
#include <hip/hip_runtime.h>

// ---------------- problem constants (match reference) ----------------
#define N_  50000
#define E_  800000
#define T_  3
#define H_  48
#define FO_ 16
#define G_  64
#define NWAVES 8

typedef __attribute__((ext_vector_type(16))) __bf16 v16bf;
typedef __attribute__((ext_vector_type(8)))  __bf16 v8bf;
typedef __attribute__((ext_vector_type(4)))  __bf16 v4bf;
typedef __attribute__((ext_vector_type(8)))  float  v8f;

// order-preserving float <-> uint key for integer atomicMax/Min
__device__ __forceinline__ unsigned f2key(float f) {
    unsigned u = __float_as_uint(f);
    return (u & 0x80000000u) ? ~u : (u | 0x80000000u);
}
__device__ __forceinline__ float key2f(unsigned k) {
    return __uint_as_float((k & 0x80000000u) ? (k & 0x7FFFFFFFu) : ~k);
}

__device__ __forceinline__ v16bf join16(v8bf lo, v8bf hi) {
    v16bf r;
#pragma unroll
    for (int j = 0; j < 8; ++j) { r[j] = lo[j]; r[j + 8] = hi[j]; }
    return r;
}

// ---------------- utility fills ----------------
__global__ void fill_f32(float* p, float v, long long n) {
    long long i = (long long)blockIdx.x * blockDim.x + threadIdx.x;
    if (i < n) p[i] = v;
}
__global__ void fill_u32(unsigned* p, unsigned v, long long n) {
    long long i = (long long)blockIdx.x * blockDim.x + threadIdx.x;
    if (i < n) p[i] = v;
}

// ---------------- weight pre-conversion to bf16 fragment-major layouts ----------------
// pre_W [T][2FIN][FIN] -> wt[(t*FIN+n)*2FIN + K]
__global__ void prep_preW(const float* __restrict__ w, __bf16* __restrict__ wt, int FIN) {
    int K2 = 2 * FIN;
    int total = T_ * K2 * FIN;
    int i = blockIdx.x * blockDim.x + threadIdx.x;
    if (i >= total) return;
    int n = i % FIN, K = (i / FIN) % K2, t = i / (FIN * K2);
    wt[((size_t)t * FIN + n) * K2 + K] = (__bf16)w[i];
}
// post_W [T][K16][16] -> wt[(t*16+n)*K16 + K]
__global__ void prep_postW(const float* __restrict__ w, __bf16* __restrict__ wt, int K16) {
    int total = T_ * K16 * FO_;
    int i = blockIdx.x * blockDim.x + threadIdx.x;
    if (i >= total) return;
    int n = i % FO_, K = (i / FO_) % K16, t = i / (FO_ * K16);
    wt[((size_t)t * FO_ + n) * K16 + K] = (__bf16)w[i];
}
// lin_W [48][48] -> wt[n*64 + K], zero-padded K=48..63
__global__ void prep_linW(const float* __restrict__ w, __bf16* __restrict__ wt) {
    int i = blockIdx.x * blockDim.x + threadIdx.x;
    if (i >= H_ * 64) return;
    int K = i % 64, n = i / 64;
    wt[i] = (K < H_) ? (__bf16)w[K * H_ + n] : (__bf16)0.f;
}

// ---------------- degrees + avg_log ----------------
__global__ void deg_kernel(const int* __restrict__ dst, float* __restrict__ cnt) {
    long long e = (long long)blockIdx.x * blockDim.x + threadIdx.x;
    if (e < E_) atomicAdd(&cnt[dst[e]], 1.0f);
}
__global__ void avglog_kernel(const float* __restrict__ cnt, float* __restrict__ scal) {
    __shared__ float red[256];
    float s = 0.f;
    for (int i = threadIdx.x; i < N_; i += 256) s += logf(cnt[i] + 1.0f);
    red[threadIdx.x] = s; __syncthreads();
    for (int k = 128; k > 0; k >>= 1) {
        if (threadIdx.x < k) red[threadIdx.x] += red[threadIdx.x + k];
        __syncthreads();
    }
    if (threadIdx.x == 0) scal[0] = red[0] / (float)N_;
}

// ---------------- edge message GEMM (WMMA) fused with 5-way scatter-agg ----------------
// one wave32 == one 16-edge tile; m = cat(x[dst],x[src]) @ pre_W[t] + pre_b
template <int FIN>
__global__ __launch_bounds__(256) void edge_kernel(
    const float* __restrict__ x, int xstride,
    const int* __restrict__ src, const int* __restrict__ dst,
    const __bf16* __restrict__ preWT, const float* __restrict__ preb,
    float* __restrict__ sum, float* __restrict__ sumsq,
    unsigned* __restrict__ mxk, unsigned* __restrict__ mnk)
{
    constexpr int K2  = 2 * FIN;
    constexpr int C4  = K2 / 4;
    constexpr int NKT = K2 / 32;
    __shared__ __align__(16) __bf16 lh[NWAVES][16][K2];
    __shared__ int ldst[NWAVES][16];
    const int lane = threadIdx.x & 31;
    const int wv   = threadIdx.x >> 5;
    const long long tileBase = ((long long)blockIdx.x * NWAVES + wv) * 16;

    // prefetch the index stream a few tiles ahead (global_prefetch_b8)
    if (lane == 0 && tileBase + 4096 < E_) {
        __builtin_prefetch(dst + tileBase + 4096, 0, 0);
        __builtin_prefetch(src + tileBase + 4096, 0, 0);
    }
    if (lane < 16) {
        long long e = tileBase + lane;
        ldst[wv][lane] = (e < E_) ? dst[e] : 0;
    }
    // stage gather tile [16 edges][2*FIN] as bf16 in LDS (float4 gathers, b64 LDS stores)
    for (int i = lane; i < 16 * C4; i += 32) {
        int row = i / C4, c4 = (i % C4) * 4;
        long long e = tileBase + row;
        v4bf q;
        if (e < E_) {
            int node = (c4 < FIN) ? dst[e] : src[e];
            int cc = (c4 < FIN) ? c4 : c4 - FIN;
            float4 f = *(const float4*)(x + (size_t)node * xstride + cc);
            q[0] = (__bf16)f.x; q[1] = (__bf16)f.y; q[2] = (__bf16)f.z; q[3] = (__bf16)f.w;
        } else {
            q[0] = q[1] = q[2] = q[3] = (__bf16)0.f;
        }
        *(v4bf*)&lh[wv][row][c4] = q;
    }
    __syncthreads();

    const int hi  = lane >> 4;      // half-wave select
    const int nl  = lane & 15;
    const int kbA = hi * 16;
    const int kbB = hi * 16;

    // A fragments depend only on kt: load once, reuse across all (tower, n-tile)
    v16bf afr[NKT];
#pragma unroll
    for (int kt = 0; kt < NKT; ++kt) {
        const __bf16* ap = &lh[wv][nl][kt * 32 + kbA];
        afr[kt] = join16(*(const v8bf*)ap, *(const v8bf*)(ap + 8));
    }
    // per-row scatter bases
    size_t obase[8];
    bool   oval[8];
#pragma unroll
    for (int r = 0; r < 8; ++r) {
        long long e = tileBase + r + hi * 8;
        oval[r]  = (e < E_);
        obase[r] = (size_t)ldst[wv][r + hi * 8] * (T_ * FIN);
    }

    for (int t = 0; t < T_; ++t) {
        for (int nt = 0; nt < FIN / 16; ++nt) {
            v8f c = {};
#pragma unroll
            for (int kt = 0; kt < NKT; ++kt) {
                // B fragment: one contiguous 32B run from fragment-major weights
                v16bf b = *(const v16bf*)(preWT +
                            ((size_t)t * FIN + nt * 16 + nl) * K2 + kt * 32 + kbB);
                c = __builtin_amdgcn_wmma_f32_16x16x32_bf16(
                        false, afr[kt], false, b, (short)0, c, false, false);
            }
            float bias = preb[t * FIN + nt * 16 + nl];
            const int fofs = t * FIN + nt * 16 + nl;
#pragma unroll
            for (int r = 0; r < 8; ++r) {
                if (oval[r]) {
                    float v = c[r] + bias;
                    size_t o = obase[r] + fofs;
                    atomicAdd(&sum[o], v);
                    atomicAdd(&sumsq[o], v * v);
                    atomicMax(&mxk[o], f2key(v));
                    atomicMin(&mnk[o], f2key(v));
                }
            }
        }
    }
}

// ---------------- build post_in = cat([x, agg*3scalers]) as bf16 ----------------
template <int FIN>
__global__ void postin_kernel(
    const float* __restrict__ x, int xstride,
    const float* __restrict__ sum, const float* __restrict__ sumsq,
    const unsigned* __restrict__ mxk, const unsigned* __restrict__ mnk,
    const float* __restrict__ cnt, const float* __restrict__ scal,
    __bf16* __restrict__ postin)
{
    constexpr int K16 = 16 * FIN;
    long long idx = (long long)blockIdx.x * blockDim.x + threadIdx.x;
    long long total = (long long)N_ * T_ * K16;
    if (idx >= total) return;
    int c = (int)(idx % K16);
    int t = (int)((idx / K16) % T_);
    long long n = idx / ((long long)K16 * T_);
    float v;
    if (c < FIN) {
        v = x[(size_t)n * xstride + c];
    } else {
        int cc = c - FIN;
        int sc = cc / (5 * FIN);        // scaler: identity / amplify / attenuate
        int a  = cc % (5 * FIN);
        int ag = a / FIN;               // sum, mean, std, max, min
        int f  = a % FIN;
        size_t o = (size_t)n * (T_ * FIN) + t * FIN + f;
        float ct  = cnt[n];
        float ctc = fmaxf(ct, 1.f);
        float inv = 1.f / ctc;
        float s   = sum[o];
        float base;
        if (ag == 0)      base = s;
        else if (ag == 1) base = s * inv;
        else if (ag == 2) {
            float mean = s * inv, m2 = sumsq[o] * inv;
            base = sqrtf(fmaxf(m2 - mean * mean, 0.f) + 1e-5f);
        }
        else if (ag == 3) base = (ct > 0.f) ? key2f(mxk[o]) : 0.f;
        else              base = (ct > 0.f) ? key2f(mnk[o]) : 0.f;
        float avg_log = scal[0];
        float ds = logf(ctc + 1.f);
        float scale = (sc == 0) ? 1.f : ((sc == 1) ? ds / avg_log : avg_log / ds);
        v = base * scale;
    }
    postin[idx] = (__bf16)v;
}

// ---------------- per-tower post GEMM (WMMA): [16 nodes x 16F] x [16F x 16] ----------------
template <int FIN>
__global__ __launch_bounds__(256) void postgemm_kernel(
    const __bf16* __restrict__ postin, const __bf16* __restrict__ postWT,
    const float* __restrict__ postb, float* __restrict__ y)
{
    constexpr int K16 = 16 * FIN;
    const int ntiles = (N_ + 15) / 16;
    int gw = blockIdx.x * NWAVES + (threadIdx.x >> 5);
    if (gw >= ntiles * T_) return;            // wave-uniform exit
    int tile = gw / T_, t = gw % T_;
    int lane = threadIdx.x & 31, nl = lane & 15, hi = lane >> 4;
    int kbA = hi * 16, kbB = hi * 16;
    int nodeA = tile * 16 + nl; if (nodeA >= N_) nodeA = N_ - 1;

    v8f c = {};
    for (int kt = 0; kt < K16 / 32; ++kt) {
        const __bf16* ap = postin + ((size_t)nodeA * T_ + t) * K16 + kt * 32 + kbA;
        v16bf a = join16(*(const v8bf*)ap, *(const v8bf*)(ap + 8));
        v16bf b = *(const v16bf*)(postWT + ((size_t)t * FO_ + nl) * K16 + kt * 32 + kbB);
        c = __builtin_amdgcn_wmma_f32_16x16x32_bf16(
                false, a, false, b, (short)0, c, false, false);
    }
    float bias = postb[t * FO_ + nl];
#pragma unroll
    for (int r = 0; r < 8; ++r) {
        int node = tile * 16 + r + hi * 8;
        if (node < N_) y[(size_t)node * H_ + t * FO_ + nl] = c[r] + bias;
    }
}

// ---------------- mixing Linear 48x48 (WMMA, K padded 48->64) ----------------
__global__ __launch_bounds__(256) void lin_kernel(
    const float* __restrict__ y, const __bf16* __restrict__ linWT,
    const float* __restrict__ linb, float* __restrict__ z)
{
    const int ntiles = (N_ + 15) / 16;
    int gw = blockIdx.x * NWAVES + (threadIdx.x >> 5);
    if (gw >= ntiles) return;
    int tile = gw;
    int lane = threadIdx.x & 31, nl = lane & 15, hi = lane >> 4;
    int kbA = hi * 16, kbB = hi * 16;
    int nodeA = tile * 16 + nl; if (nodeA >= N_) nodeA = N_ - 1;

    // A fragments (K padded 48->64) shared across the three n-tiles
    v16bf afr[2];
#pragma unroll
    for (int kt = 0; kt < 2; ++kt) {
        v16bf a;
#pragma unroll
        for (int j = 0; j < 16; ++j) {
            int K = kt * 32 + kbA + j;
            a[j] = (K < H_) ? (__bf16)y[(size_t)nodeA * H_ + K] : (__bf16)0.f;
        }
        afr[kt] = a;
    }

    for (int nt = 0; nt < H_ / 16; ++nt) {
        v8f c = {};
#pragma unroll
        for (int kt = 0; kt < 2; ++kt) {
            v16bf b = *(const v16bf*)(linWT + (size_t)(nt * 16 + nl) * 64 + kt * 32 + kbB);
            c = __builtin_amdgcn_wmma_f32_16x16x32_bf16(
                    false, afr[kt], false, b, (short)0, c, false, false);
        }
        float bias = linb[nt * 16 + nl];
#pragma unroll
        for (int r = 0; r < 8; ++r) {
            int node = tile * 16 + r + hi * 8;
            if (node < N_) z[(size_t)node * H_ + nt * 16 + nl] = c[r] + bias;
        }
    }
}

// ---------------- BatchNorm stats (one block per feature, no atomics) ----------------
__global__ void bnstat_kernel(const float* __restrict__ z,
                              float* __restrict__ bnsum, float* __restrict__ bnsumsq)
{
    int c = blockIdx.x;
    __shared__ float r1[256], r2[256];
    float s = 0.f, s2 = 0.f;
    for (int n = threadIdx.x; n < N_; n += 256) {
        float v = z[(size_t)n * H_ + c];
        s += v; s2 += v * v;
    }
    r1[threadIdx.x] = s; r2[threadIdx.x] = s2; __syncthreads();
    for (int k = 128; k > 0; k >>= 1) {
        if (threadIdx.x < k) { r1[threadIdx.x] += r1[threadIdx.x + k]; r2[threadIdx.x] += r2[threadIdx.x + k]; }
        __syncthreads();
    }
    if (threadIdx.x == 0) { bnsum[c] = r1[0]; bnsumsq[c] = r2[0]; }
}

__global__ void bnapply_kernel(const float* __restrict__ z,
                               const float* __restrict__ bnsum, const float* __restrict__ bnsumsq,
                               const float* __restrict__ g, const float* __restrict__ b,
                               float* __restrict__ h)
{
    long long idx = (long long)blockIdx.x * blockDim.x + threadIdx.x;
    if (idx >= (long long)N_ * H_) return;
    int c = (int)(idx % H_);
    float mu  = bnsum[c] / (float)N_;
    float var = bnsumsq[c] / (float)N_ - mu * mu;
    float v = (z[idx] - mu) * rsqrtf(var + 1e-5f) * g[c] + b[c];
    h[idx] = fmaxf(v, 0.f);
}

// ---------------- graph mean-pool + head ----------------
__global__ void pool_kernel(const float* __restrict__ h, const int* __restrict__ batch,
                            float* __restrict__ pooled, float* __restrict__ gcnt)
{
    long long idx = (long long)blockIdx.x * blockDim.x + threadIdx.x;
    if (idx >= (long long)N_ * H_) return;
    int n = (int)(idx / H_), c = (int)(idx % H_);
    int g = batch[n];
    atomicAdd(&pooled[g * H_ + c], h[idx]);
    if (c == 0) atomicAdd(&gcnt[g], 1.0f);
}

__global__ void final_kernel(const float* __restrict__ pooled, const float* __restrict__ gcnt,
                             const float* __restrict__ outW, const float* __restrict__ outb,
                             float* __restrict__ out)
{
    int g = threadIdx.x;
    if (g < G_) {
        float inv = 1.f / fmaxf(gcnt[g], 1.f);
        float acc = outb[0];
        for (int c = 0; c < H_; ++c) acc += pooled[g * H_ + c] * inv * outW[c];
        out[g] = acc;
    }
}

// ---------------- host launch ----------------
static inline unsigned grd(long long n) { return (unsigned)((n + 255) / 256); }

extern "C" void kernel_launch(void* const* d_in, const int* in_sizes, int n_in,
                              void* d_out, int out_size, void* d_ws, size_t ws_size,
                              hipStream_t stream) {
    const float* x0    = (const float*)d_in[0];
    const int*   ei    = (const int*)d_in[1];
    const int*   src   = ei;
    const int*   dst   = ei + E_;
    const int*   batch = (const int*)d_in[2];
    const float *preW[3], *preb[3], *postW[3], *postb[3], *linW[3], *linb[3], *bng[3], *bnb[3];
    for (int l = 0; l < 3; ++l) {
        preW[l]  = (const float*)d_in[3 + 8 * l + 0];
        preb[l]  = (const float*)d_in[3 + 8 * l + 1];
        postW[l] = (const float*)d_in[3 + 8 * l + 2];
        postb[l] = (const float*)d_in[3 + 8 * l + 3];
        linW[l]  = (const float*)d_in[3 + 8 * l + 4];
        linb[l]  = (const float*)d_in[3 + 8 * l + 5];
        bng[l]   = (const float*)d_in[3 + 8 * l + 6];
        bnb[l]   = (const float*)d_in[3 + 8 * l + 7];
    }
    const float* outW = (const float*)d_in[27];
    const float* outb = (const float*)d_in[28];

    // carve workspace
    char* ws = (char*)d_ws;
    size_t off = 0;
    auto carve = [&](size_t bytes) -> char* {
        char* p = ws + off;
        off = (off + bytes + 255) & ~(size_t)255;
        return p;
    };
    float*    cnt    = (float*)carve((size_t)N_ * 4);
    float*    scal   = (float*)carve(256);
    float*    hbuf   = (float*)carve((size_t)N_ * H_ * 4);
    float*    zbuf   = (float*)carve((size_t)N_ * H_ * 4);
    float*    ybuf   = (float*)carve((size_t)N_ * H_ * 4);
    float*    sum    = (float*)carve((size_t)N_ * T_ * H_ * 4);
    float*    sumsq  = (float*)carve((size_t)N_ * T_ * H_ * 4);
    unsigned* mxk    = (unsigned*)carve((size_t)N_ * T_ * H_ * 4);
    unsigned* mnk    = (unsigned*)carve((size_t)N_ * T_ * H_ * 4);
    __bf16*   postin = (__bf16*)carve((size_t)N_ * T_ * (16 * H_) * 2);
    float*    bnsum  = (float*)carve(H_ * 4);
    float*    bnsq   = (float*)carve(H_ * 4);
    float*    pooled = (float*)carve((size_t)G_ * H_ * 4);
    float*    gcnt   = (float*)carve((size_t)G_ * 4);
    // bf16 fragment-major weights
    __bf16* preWT[3], *postWT[3], *linWT[3];
    for (int l = 0; l < 3; ++l) {
        int FIN = (l == 0) ? 16 : 48;
        preWT[l]  = (__bf16*)carve((size_t)T_ * FIN * 2 * FIN * 2);
        postWT[l] = (__bf16*)carve((size_t)T_ * FO_ * 16 * FIN * 2);
        linWT[l]  = (__bf16*)carve((size_t)H_ * 64 * 2);
    }

    // weight prep (once per call)
    for (int l = 0; l < 3; ++l) {
        int FIN = (l == 0) ? 16 : 48;
        prep_preW <<<grd((long long)T_ * 2 * FIN * FIN), 256, 0, stream>>>(preW[l],  preWT[l],  FIN);
        prep_postW<<<grd((long long)T_ * 16 * FIN * FO_), 256, 0, stream>>>(postW[l], postWT[l], 16 * FIN);
        prep_linW <<<grd((long long)H_ * 64), 256, 0, stream>>>(linW[l], linWT[l]);
    }

    // in-degrees and avg_log (reused across layers)
    fill_f32<<<grd(N_), 256, 0, stream>>>(cnt, 0.f, N_);
    deg_kernel<<<grd(E_), 256, 0, stream>>>(dst, cnt);
    avglog_kernel<<<1, 256, 0, stream>>>(cnt, scal);

    const int etiles  = (E_ + 15) / 16;
    const int eblocks = (etiles + NWAVES - 1) / NWAVES;
    const int ntiles  = (N_ + 15) / 16;

    for (int l = 0; l < 3; ++l) {
        const int    FIN  = (l == 0) ? 16 : 48;
        const float* xin  = (l == 0) ? x0 : hbuf;
        const long long nagg = (long long)N_ * T_ * H_;

        fill_f32<<<grd(nagg), 256, 0, stream>>>(sum,   0.f, nagg);
        fill_f32<<<grd(nagg), 256, 0, stream>>>(sumsq, 0.f, nagg);
        fill_u32<<<grd(nagg), 256, 0, stream>>>(mxk, 0u,          nagg);
        fill_u32<<<grd(nagg), 256, 0, stream>>>(mnk, 0xFFFFFFFFu, nagg);

        if (l == 0) {
            edge_kernel<16><<<eblocks, 256, 0, stream>>>(
                xin, FIN, src, dst, preWT[l], preb[l], sum, sumsq, mxk, mnk);
            long long npi = (long long)N_ * T_ * (16 * 16);
            postin_kernel<16><<<grd(npi), 256, 0, stream>>>(
                xin, FIN, sum, sumsq, mxk, mnk, cnt, scal, postin);
            int waves = ntiles * T_;
            postgemm_kernel<16><<<(waves + NWAVES - 1) / NWAVES, 256, 0, stream>>>(
                postin, postWT[l], postb[l], ybuf);
        } else {
            edge_kernel<48><<<eblocks, 256, 0, stream>>>(
                xin, FIN, src, dst, preWT[l], preb[l], sum, sumsq, mxk, mnk);
            long long npi = (long long)N_ * T_ * (16 * 48);
            postin_kernel<48><<<grd(npi), 256, 0, stream>>>(
                xin, FIN, sum, sumsq, mxk, mnk, cnt, scal, postin);
            int waves = ntiles * T_;
            postgemm_kernel<48><<<(waves + NWAVES - 1) / NWAVES, 256, 0, stream>>>(
                postin, postWT[l], postb[l], ybuf);
        }

        lin_kernel<<<(ntiles + NWAVES - 1) / NWAVES, 256, 0, stream>>>(
            ybuf, linWT[l], linb[l], zbuf);
        bnstat_kernel<<<H_, 256, 0, stream>>>(zbuf, bnsum, bnsq);
        bnapply_kernel<<<grd((long long)N_ * H_), 256, 0, stream>>>(
            zbuf, bnsum, bnsq, bng[l], bnb[l], hbuf);
    }

    fill_f32<<<grd((long long)G_ * H_), 256, 0, stream>>>(pooled, 0.f, (long long)G_ * H_);
    fill_f32<<<grd(G_), 256, 0, stream>>>(gcnt, 0.f, G_);
    pool_kernel<<<grd((long long)N_ * H_), 256, 0, stream>>>(hbuf, batch, pooled, gcnt);
    final_kernel<<<1, 64, 0, stream>>>(pooled, gcnt, outW, outb, (float*)d_out);
}
